// HydraBlock_17609365914528
// MI455X (gfx1250) — matrix operand order repaired and checked by reading, not compiled, not run on hardware
//
#include <hip/hip_runtime.h>
#include <hip/hip_bf16.h>
#include <math.h>

// ---------------------------------------------------------------------------
// HydraBlock forward for MI455X (gfx1250, wave32, WMMA + Tensor Data Mover).
// Dense layers -> v_wmma_f32_16x16x32_f16 fed from TDM-staged LDS tiles.
// Linear-attention context (K^T V, K-dim = 1024) -> WMMA + LDS cross-wave sum.
// MoE computed sparsely over the top-4 experts (mask is 0 elsewhere).
// ---------------------------------------------------------------------------

typedef __attribute__((ext_vector_type(16))) _Float16 v16h;
typedef __attribute__((ext_vector_type(8)))  float    v8f;
typedef unsigned int v4u __attribute__((ext_vector_type(4)));
typedef int          v8i __attribute__((ext_vector_type(8)));
typedef int          v4i __attribute__((ext_vector_type(4)));

#define B_   2
#define T_   1024
#define D_   128
#define H_   8
#define E_   256
#define R_   32
#define K_   4
#define NTOK (B_*T_)

// ---------------- Tensor Data Mover: 2D f32 tile -> LDS ---------------------
// D# per CDNA5 ISA ch.8: group0 = {count, lds_addr, global_addr[56:0], type=2},
// group1 = {data_size=4B, tensor_dim0/1, tile_dim0/1, tensor_dim0_stride}.
__device__ __forceinline__ void tdm_load_2d(const void* gptr, unsigned lds_off,
                                            unsigned tile_d0, unsigned tile_d1,
                                            unsigned long long row_stride,
                                            unsigned tensor_d0, unsigned tensor_d1) {
  const unsigned long long ga = (unsigned long long)(uintptr_t)gptr;
  v4u g0;
  g0[0] = 1u;                                                // count=1 (user D#)
  g0[1] = lds_off;                                           // LDS byte address
  g0[2] = (unsigned)ga;                                      // global_addr[31:0]
  g0[3] = (unsigned)((ga >> 32) & 0x01ffffffu) | (2u << 30); // [56:32] | type=2
  v8i g1;
  g1[0] = (int)(2u << 16);                                   // data_size = 4 bytes
  g1[1] = (int)((tensor_d0 & 0xffffu) << 16);                // tensor_dim0[15:0]
  g1[2] = (int)(((tensor_d0 >> 16) & 0xffffu) |
                ((tensor_d1 & 0xffffu) << 16));              // dim0[31:16]|dim1[15:0]
  g1[3] = (int)(((tensor_d1 >> 16) & 0xffffu) |
                ((tile_d0 & 0xffffu) << 16));                // dim1[31:16]|tile0
  g1[4] = (int)(tile_d1 & 0xffffu);                          // tile1 (tile2 = 0)
  g1[5] = (int)(unsigned)(row_stride & 0xffffffffull);       // dim0_stride[31:0]
  g1[6] = (int)(unsigned)((row_stride >> 32) & 0xffffull);   // dim0_stride[47:32]
  g1[7] = 0;
  v4i z4 = {0, 0, 0, 0};
#if defined(__clang_major__) && (__clang_major__ >= 23)
  v8i z8 = {0, 0, 0, 0, 0, 0, 0, 0};
  __builtin_amdgcn_tensor_load_to_lds(g0, g1, z4, z4, z8, 0);
#else
  __builtin_amdgcn_tensor_load_to_lds(g0, g1, z4, z4, 0);
#endif
}

// ---------------- TDM-staged WMMA GEMM: C[M,N] = A[M,Kd]*W[Kd,N] + bias -----
// grid = (N/64, M/16), block = 128 (4 waves share a TDM-staged 16xKd A tile
// and a Kd x 64 B tile; each wave owns one 16x16 C tile). Kd % 32 == 0, Kd<=128.
__global__ __launch_bounds__(128)
void gemm_wmma_tdm(const float* __restrict__ A, const float* __restrict__ W,
                   const float* __restrict__ bias, float* __restrict__ C,
                   int M, int N, int Kd) {
  const int tn = blockIdx.x, tm = blockIdx.y;
  const int tid = threadIdx.x;
  const int wave = tid >> 5, lane = tid & 31;
  const int hi = lane >> 4, lo = lane & 15;
  __shared__ float ldsA[16 * 128];   // A tile, row-major [16][Kd]
  __shared__ float ldsB[128 * 64];   // B tile, row-major [Kd][64]

  if (tid < 32) {  // wave 0 programs the TDM (EXEC ignored by tensor ops)
    tdm_load_2d(A + (size_t)tm * 16 * Kd, (unsigned)(uintptr_t)(void*)ldsA,
                (unsigned)Kd, 16u, (unsigned long long)Kd,
                (unsigned)Kd, (unsigned)M);
    tdm_load_2d(W + (size_t)tn * 64, (unsigned)(uintptr_t)(void*)ldsB,
                64u, (unsigned)Kd, (unsigned long long)N,
                (unsigned)N, (unsigned)Kd);
    __builtin_amdgcn_s_wait_tensorcnt(0);
  }
  __syncthreads();

  v8f acc = {};
  for (int k0 = 0; k0 < Kd; k0 += 32) {
    v16h a, bfr;
#pragma unroll
    for (int v = 0; v < 8; ++v) {          // A 16x32 f16 fragment layout
      const int kb = k0 + ((v >> 2) << 4) + (hi << 3) + ((v & 3) << 1);
      a[2*v]   = (_Float16)ldsA[lo * Kd + kb];
      a[2*v+1] = (_Float16)ldsA[lo * Kd + kb + 1];
    }
#pragma unroll
    for (int e = 0; e < 16; ++e)           // B 32x16 f16 fragment layout
      bfr[e] = (_Float16)ldsB[(k0 + ((e >> 3) << 4) + (hi << 3) + (e & 7)) * 64 + wave * 16 + lo];
    acc = __builtin_amdgcn_wmma_f32_16x16x32_f16(false, a, false, bfr,
                                                 (short)0, acc, false, false);
  }
  const int n = tn * 64 + wave * 16 + lo;
  const float bb = bias ? bias[n] : 0.f;
#pragma unroll
  for (int r = 0; r < 8; ++r) {            // C: VGPR r -> M = hi*8 + r
    const int m = tm * 16 + (hi << 3) + r;
    C[(size_t)m * N + n] = acc[r] + bb;
  }
}

// ---------------- narrow fallback GEMM (N = 32): direct-load WMMA -----------
// grid = (N/16, M/16), block = 32.
__global__ __launch_bounds__(32)
void gemm_wmma_f16(const float* __restrict__ A, const float* __restrict__ W,
                   const float* __restrict__ bias, float* __restrict__ C,
                   int M, int N, int Kd) {
  const int tn = blockIdx.x, tm = blockIdx.y;
  const int lane = threadIdx.x;
  const int hi = lane >> 4, lo = lane & 15;
  v8f acc = {};
  const float* arow = A + (size_t)(tm * 16 + lo) * Kd;
  const float* wcol = W + (tn * 16 + lo);
  for (int k0 = 0; k0 < Kd; k0 += 32) {
    v16h a, bfr;
#pragma unroll
    for (int v = 0; v < 8; ++v) {
      const int kb = k0 + ((v >> 2) << 4) + (hi << 3) + ((v & 3) << 1);
      a[2*v]   = (_Float16)arow[kb];
      a[2*v+1] = (_Float16)arow[kb + 1];
    }
#pragma unroll
    for (int e = 0; e < 16; ++e)
      bfr[e] = (_Float16)wcol[(size_t)(k0 + ((e >> 3) << 4) + (hi << 3) + (e & 7)) * N];
    acc = __builtin_amdgcn_wmma_f32_16x16x32_f16(false, a, false, bfr,
                                                 (short)0, acc, false, false);
  }
  const int n = tn * 16 + lo;
  const float bb = bias ? bias[n] : 0.f;
#pragma unroll
  for (int r = 0; r < 8; ++r) {
    const int m = tm * 16 + (hi << 3) + r;
    C[(size_t)m * N + n] = acc[r] + bb;
  }
}

// ---------------- rmsnorm: one wave per token, 4 floats / lane --------------
__global__ __launch_bounds__(32)
void rmsnorm_kernel(const float* __restrict__ x, const float* __restrict__ gamma,
                    float* __restrict__ out) {
  const int tok = blockIdx.x, lane = threadIdx.x;
  const float4 v = reinterpret_cast<const float4*>(x + (size_t)tok * D_)[lane];
  float ss = v.x*v.x + v.y*v.y + v.z*v.z + v.w*v.w;
#pragma unroll
  for (int off = 16; off > 0; off >>= 1) ss += __shfl_xor(ss, off, 32);
  const float sc = rsqrtf(ss * (1.f / 128.f) + 1e-6f);
  const float4 g = reinterpret_cast<const float4*>(gamma)[lane];
  float4 o; o.x = v.x*sc*g.x; o.y = v.y*sc*g.y; o.z = v.z*sc*g.z; o.w = v.w*sc*g.w;
  reinterpret_cast<float4*>(out + (size_t)tok * D_)[lane] = o;
}

// ---------------- RoPE + elu+1 for q,k; extract v ---------------------------
__global__ __launch_bounds__(128)
void rope_kernel(const float* __restrict__ qlin, const float* __restrict__ kvu,
                 float* __restrict__ qp, float* __restrict__ kp,
                 float* __restrict__ vb) {
  const int tok = blockIdx.x;
  const int t = tok & (T_ - 1);
  const int tid = threadIdx.x;
  const int h = tid >> 4, d = tid & 15;
  const int j = d & 7;
  const float ang = (float)t * __powf(10000.f, -(float)j * 0.125f);
  const float c = __cosf(ang), s = __sinf(ang);
  {
    const float x1 = qlin[(size_t)tok * D_ + h * 16 + j];
    const float x2 = qlin[(size_t)tok * D_ + h * 16 + 8 + j];
    const float o = (d < 8) ? (x1 * c - x2 * s) : (x1 * s + x2 * c);
    qp[(size_t)tok * D_ + tid] = (o > 0.f) ? (o + 1.f) : __expf(o);  // elu+1
  }
  {
    const float x1 = kvu[(size_t)tok * (2 * D_) + h * 32 + j];
    const float x2 = kvu[(size_t)tok * (2 * D_) + h * 32 + 8 + j];
    const float o = (d < 8) ? (x1 * c - x2 * s) : (x1 * s + x2 * c);
    kp[(size_t)tok * D_ + tid] = (o > 0.f) ? (o + 1.f) : __expf(o);
    vb[(size_t)tok * D_ + tid] = kvu[(size_t)tok * (2 * D_) + h * 32 + 16 + d];
  }
}

// ---------------- ctx[bh] = K^T V (16x16, K-dim=1024) via WMMA --------------
__global__ __launch_bounds__(256)
void ctx_kernel(const float* __restrict__ kp, const float* __restrict__ vbuf,
                float* __restrict__ ctx, float* __restrict__ ksum) {
  const int bh = blockIdx.x;
  const int b = bh >> 3, h = bh & 7;
  const int tid = threadIdx.x;
  const int wave = tid >> 5, lane = tid & 31;
  const int hi = lane >> 4, lo = lane & 15;
  __shared__ float part[8 * 256];
  __shared__ float ks[16 * 16];

  const int tokbase = b * T_ + wave * 128;
  v8f acc = {};
  for (int c = 0; c < 4; ++c) {
    const int t0 = tokbase + c * 32;
    v16h a, bfr;
#pragma unroll
    for (int v = 0; v < 8; ++v) {          // A[d, t] = k[t, d]
      const int kb = ((v >> 2) << 4) + (hi << 3) + ((v & 3) << 1);
      a[2*v]   = (_Float16)kp[(size_t)(t0 + kb    ) * D_ + h * 16 + lo];
      a[2*v+1] = (_Float16)kp[(size_t)(t0 + kb + 1) * D_ + h * 16 + lo];
    }
#pragma unroll
    for (int e = 0; e < 16; ++e)           // B[t, e] = v[t, e]
      bfr[e] = (_Float16)vbuf[(size_t)(t0 + ((e >> 3) << 4) + (hi << 3) + (e & 7)) * D_ + h * 16 + lo];
    acc = __builtin_amdgcn_wmma_f32_16x16x32_f16(false, a, false, bfr,
                                                 (short)0, acc, false, false);
  }
#pragma unroll
  for (int r = 0; r < 8; ++r) part[wave * 256 + r * 32 + lane] = acc[r];

  {  // k_sum partials: 16 chunks x 64 tokens
    const int d = tid & 15, chunk = tid >> 4;
    float s = 0.f;
    const int tb = b * T_ + chunk * 64;
    for (int t = 0; t < 64; ++t) s += kp[(size_t)(tb + t) * D_ + h * 16 + d];
    ks[chunk * 16 + d] = s;
  }
  __syncthreads();
  {
    float tot = 0.f;
#pragma unroll
    for (int w = 0; w < 8; ++w) tot += part[w * 256 + tid];
    const int r = tid >> 5, ln = tid & 31;
    const int d = ((ln >> 4) << 3) + r, e = ln & 15;
    ctx[(size_t)bh * 256 + d * 16 + e] = tot;
  }
  if (tid < 16) {
    float s = 0.f;
#pragma unroll
    for (int c = 0; c < 16; ++c) s += ks[c * 16 + tid];
    ksum[bh * 16 + tid] = s;
  }
}

// ---------------- out[t, h*16+e] = (q . ctx[:,e]) / (q . ksum + 1e-6) -------
__global__ __launch_bounds__(128)
void attn_apply_kernel(const float* __restrict__ qp, const float* __restrict__ ctx,
                       const float* __restrict__ ksum, float* __restrict__ out) {
  const int tok = blockIdx.x;
  const int b = tok >> 10;
  const int tid = threadIdx.x, h = tid >> 4, e = tid & 15;
  const int bh = b * H_ + h;
  float num = 0.f, den = 0.f;
#pragma unroll
  for (int d = 0; d < 16; ++d) {
    const float q = qp[(size_t)tok * D_ + h * 16 + d];
    num = fmaf(q, ctx[(size_t)bh * 256 + d * 16 + e], num);
    den = fmaf(q, ksum[bh * 16 + d], den);
  }
  out[(size_t)tok * D_ + tid] = num / (den + 1e-6f);
}

// ---------------- TurboQuant: one wave per token ----------------------------
__global__ __launch_bounds__(32)
void turbo_kernel(const float* __restrict__ r, const float* __restrict__ scale,
                  float* __restrict__ out) {
  const int tok = blockIdx.x, lane = threadIdx.x;
  const float4 v = reinterpret_cast<const float4*>(r + (size_t)tok * D_)[lane];
  float ss = v.x*v.x + v.y*v.y + v.z*v.z + v.w*v.w;
#pragma unroll
  for (int off = 16; off > 0; off >>= 1) ss += __shfl_xor(ss, off, 32);
  const float mag = sqrtf(ss * (1.f / 128.f) + 1e-6f);
  const float inv = 1.f / mag;
  const float4 g = reinterpret_cast<const float4*>(scale)[lane];
  float4 o;
  o.x = fminf(fmaxf(v.x*inv*127.f, -127.f), 127.f) * (1.f/127.f) * mag * g.x;
  o.y = fminf(fmaxf(v.y*inv*127.f, -127.f), 127.f) * (1.f/127.f) * mag * g.y;
  o.z = fminf(fmaxf(v.z*inv*127.f, -127.f), 127.f) * (1.f/127.f) * mag * g.z;
  o.w = fminf(fmaxf(v.w*inv*127.f, -127.f), 127.f) * (1.f/127.f) * mag * g.w;
  reinterpret_cast<float4*>(out + (size_t)tok * D_)[lane] = o;
}

// ---------------- SwiGLU combine (optional residual base) -------------------
__global__ void swiglu_kernel(const float* __restrict__ g, const float* __restrict__ u,
                              const float* __restrict__ base, float* __restrict__ out,
                              int n) {
  const int i = blockIdx.x * blockDim.x + threadIdx.x;
  if (i < n) {
    const float gv = g[i];
    const float s = gv / (1.f + __expf(-gv)) * u[i];
    out[i] = base ? (base[i] + s) : s;
  }
}

// ---------------- softmax + top-4 router: one wave per token ----------------
__global__ __launch_bounds__(32)
void topk_kernel(const float* __restrict__ gs, int* __restrict__ topi,
                 float* __restrict__ topw) {
  const int tok = blockIdx.x, lane = threadIdx.x;
  float s[8];
#pragma unroll
  for (int i = 0; i < 8; ++i) s[i] = gs[(size_t)tok * E_ + lane + 32 * i];
  float mx = s[0];
#pragma unroll
  for (int i = 1; i < 8; ++i) mx = fmaxf(mx, s[i]);
  for (int off = 16; off > 0; off >>= 1) mx = fmaxf(mx, __shfl_xor(mx, off, 32));
  float zs = 0.f;
#pragma unroll
  for (int i = 0; i < 8; ++i) zs += __expf(s[i] - mx);
  for (int off = 16; off > 0; off >>= 1) zs += __shfl_xor(zs, off, 32);

  unsigned picked = 0u;
  float pv[4]; int pi[4];
  for (int j = 0; j < 4; ++j) {
    float bv = -1e30f; int bi = E_;
#pragma unroll
    for (int i = 0; i < 8; ++i)
      if (!(picked & (1u << i)) && s[i] > bv) { bv = s[i]; bi = lane + 32 * i; }
    for (int off = 16; off > 0; off >>= 1) {
      const float ov = __shfl_xor(bv, off, 32);
      const int   oi = __shfl_xor(bi, off, 32);
      if (ov > bv || (ov == bv && oi < bi)) { bv = ov; bi = oi; }
    }
    if ((bi & 31) == lane) picked |= 1u << (bi >> 5);
    pv[j] = __expf(bv - mx) / zs;
    pi[j] = bi;
  }
  if (lane == 0) {
    const float ws = pv[0] + pv[1] + pv[2] + pv[3] + 1e-6f;
#pragma unroll
    for (int j = 0; j < 4; ++j) { topi[tok*4+j] = pi[j]; topw[tok*4+j] = pv[j] / ws; }
  }
}

// ---------------- sparse MoE: only the top-4 experts matter -----------------
__global__ __launch_bounds__(128)
void expert_kernel(const float* __restrict__ h2, const float* __restrict__ ew,
                   const int* __restrict__ topi, const float* __restrict__ topw,
                   float* __restrict__ eo, float* __restrict__ mixed) {
  const int tok = blockIdx.x, d = threadIdx.x;
  __shared__ float hs[D_];
  hs[d] = h2[(size_t)tok * D_ + d];
  __syncthreads();
  float mix = 0.f;
#pragma unroll
  for (int j = 0; j < K_; ++j) {
    const int e = topi[tok * 4 + j];
    const float w = topw[tok * 4 + j];
    if (j + 1 < K_) {  // hint next expert panel toward L2 (global_prefetch_b8)
      const int en = topi[tok * 4 + j + 1];
      __builtin_prefetch(ew + (size_t)en * D_ * D_ + d, 0, 1);
    }
    const float* wp = ew + (size_t)e * D_ * D_ + d;
    float acc = 0.f;
#pragma unroll 8
    for (int kk = 0; kk < D_; ++kk) acc = fmaf(hs[kk], wp[(size_t)kk * D_], acc);
    eo[((size_t)tok * 4 + j) * D_ + d] = acc;
    mix = fmaf(w, acc, mix);
  }
  mixed[(size_t)tok * D_ + d] = mix;
}

// ---------------- residual + weighted variance -> consensus -----------------
__global__ __launch_bounds__(32)
void final_kernel(const float* __restrict__ x1, const float* __restrict__ wa,
                  const float* __restrict__ eo, const float* __restrict__ topw,
                  float* __restrict__ xout, float* __restrict__ consensus) {
  const int tok = blockIdx.x, lane = threadIdx.x;
  float tw[4];
#pragma unroll
  for (int j = 0; j < 4; ++j) tw[j] = topw[tok * 4 + j];
  float vsum = 0.f;
#pragma unroll
  for (int q = 0; q < 4; ++q) {
    const int d = lane * 4 + q;
    const float w = wa[(size_t)tok * D_ + d];
    float vd = 0.f;
#pragma unroll
    for (int j = 0; j < 4; ++j) {
      const float df = eo[((size_t)tok * 4 + j) * D_ + d] - w;
      vd = fmaf(tw[j], df * df, vd);
    }
    vsum += vd;
    xout[(size_t)tok * D_ + d] = x1[(size_t)tok * D_ + d] + w;
  }
  for (int off = 16; off > 0; off >>= 1) vsum += __shfl_xor(vsum, off, 32);
  if (lane == 0) consensus[tok] = __expf(-vsum * (1.f / 128.f));
}

// ---------------------------------------------------------------------------
extern "C" void kernel_launch(void* const* d_in, const int* in_sizes, int n_in,
                              void* d_out, int out_size, void* d_ws, size_t ws_size,
                              hipStream_t stream) {
  (void)in_sizes; (void)n_in; (void)out_size; (void)ws_size;
  const float* x        = (const float*)d_in[0];
  const float* n1_gamma = (const float*)d_in[1];
  const float* q_w      = (const float*)d_in[2];
  const float* q_b      = (const float*)d_in[3];
  const float* kvd_w    = (const float*)d_in[4];
  const float* kvd_b    = (const float*)d_in[5];
  const float* kvu_w    = (const float*)d_in[6];
  const float* kvu_b    = (const float*)d_in[7];
  const float* out_w    = (const float*)d_in[8];
  const float* out_b    = (const float*)d_in[9];
  const float* tq_rot   = (const float*)d_in[10];
  const float* tq_scale = (const float*)d_in[11];
  const float* sg1_w1   = (const float*)d_in[12];
  const float* sg1_b1   = (const float*)d_in[13];
  const float* sg1_w2   = (const float*)d_in[14];
  const float* sg1_b2   = (const float*)d_in[15];
  const float* n2_gamma = (const float*)d_in[16];
  const float* gate_w   = (const float*)d_in[17];
  const float* gate_b   = (const float*)d_in[18];
  const float* expert_w = (const float*)d_in[19];
  const float* sg2_w1   = (const float*)d_in[20];
  const float* sg2_b1   = (const float*)d_in[21];
  const float* sg2_w2   = (const float*)d_in[22];
  const float* sg2_b2   = (const float*)d_in[23];

  char* wsp = (char*)d_ws;
  auto carve = [&](size_t nbytes) -> void* {
    void* p = (void*)wsp;
    wsp += (nbytes + 255) & ~(size_t)255;
    return p;
  };
  const size_t TD = (size_t)NTOK * D_ * sizeof(float);
  float* h      = (float*)carve(TD);
  float* q_lin  = (float*)carve(TD);
  float* kvd    = (float*)carve((size_t)NTOK * R_ * sizeof(float));
  float* kvu    = (float*)carve((size_t)NTOK * 2 * D_ * sizeof(float));
  float* qp     = (float*)carve(TD);
  float* kp     = (float*)carve(TD);
  float* vbuf   = (float*)carve(TD);
  float* ctx    = (float*)carve((size_t)B_ * H_ * 256 * sizeof(float));
  float* ksum   = (float*)carve((size_t)B_ * H_ * 16 * sizeof(float));
  float* attnin = (float*)carve(TD);
  float* attnp  = (float*)carve(TD);
  float* rbuf   = (float*)carve(TD);
  float* tqbuf  = (float*)carve(TD);
  float* g1     = (float*)carve(TD);
  float* u1     = (float*)carve(TD);
  float* x1     = (float*)carve(TD);
  float* h2     = (float*)carve(TD);
  float* gs     = (float*)carve((size_t)NTOK * E_ * sizeof(float));
  int*   topi   = (int*)  carve((size_t)NTOK * K_ * sizeof(int));
  float* topw   = (float*)carve((size_t)NTOK * K_ * sizeof(float));
  float* eo     = (float*)carve((size_t)NTOK * K_ * D_ * sizeof(float));
  float* mixed  = (float*)carve(TD);
  float* g2     = (float*)carve(TD);
  float* u2     = (float*)carve(TD);
  float* wa     = (float*)carve(TD);

  float* xout = (float*)d_out;
  float* cons = xout + (size_t)NTOK * D_;

  const dim3 w32(32), b128(128), b256(256);

  // ---- attention branch ----
  rmsnorm_kernel<<<NTOK, w32, 0, stream>>>(x, n1_gamma, h);
  gemm_wmma_tdm<<<dim3(D_/64, NTOK/16), b128, 0, stream>>>(h, q_w, q_b, q_lin, NTOK, D_, D_);
  gemm_wmma_f16<<<dim3(R_/16, NTOK/16), w32, 0, stream>>>(h, kvd_w, kvd_b, kvd, NTOK, R_, D_);
  gemm_wmma_tdm<<<dim3(2*D_/64, NTOK/16), b128, 0, stream>>>(kvd, kvu_w, kvu_b, kvu, NTOK, 2*D_, R_);
  rope_kernel<<<NTOK, b128, 0, stream>>>(q_lin, kvu, qp, kp, vbuf);
  ctx_kernel<<<B_*H_, b256, 0, stream>>>(kp, vbuf, ctx, ksum);
  attn_apply_kernel<<<NTOK, b128, 0, stream>>>(qp, ctx, ksum, attnin);
  gemm_wmma_tdm<<<dim3(D_/64, NTOK/16), b128, 0, stream>>>(attnin, out_w, out_b, attnp, NTOK, D_, D_);

  // ---- TurboQuant ----
  gemm_wmma_tdm<<<dim3(D_/64, NTOK/16), b128, 0, stream>>>(attnp, tq_rot, nullptr, rbuf, NTOK, D_, D_);
  turbo_kernel<<<NTOK, w32, 0, stream>>>(rbuf, tq_scale, tqbuf);

  // ---- SwiGLU residual ----
  gemm_wmma_tdm<<<dim3(D_/64, NTOK/16), b128, 0, stream>>>(tqbuf, sg1_w1, sg1_b1, g1, NTOK, D_, D_);
  gemm_wmma_tdm<<<dim3(D_/64, NTOK/16), b128, 0, stream>>>(tqbuf, sg1_w2, sg1_b2, u1, NTOK, D_, D_);
  swiglu_kernel<<<(NTOK*D_ + 255)/256, b256, 0, stream>>>(g1, u1, x, x1, NTOK*D_);

  // ---- MoE ----
  rmsnorm_kernel<<<NTOK, w32, 0, stream>>>(x1, n2_gamma, h2);
  gemm_wmma_tdm<<<dim3(E_/64, NTOK/16), b128, 0, stream>>>(h2, gate_w, gate_b, gs, NTOK, E_, D_);
  topk_kernel<<<NTOK, w32, 0, stream>>>(gs, topi, topw);
  expert_kernel<<<NTOK, b128, 0, stream>>>(h2, expert_w, topi, topw, eo, mixed);
  gemm_wmma_tdm<<<dim3(D_/64, NTOK/16), b128, 0, stream>>>(mixed, sg2_w1, sg2_b1, g2, NTOK, D_, D_);
  gemm_wmma_tdm<<<dim3(D_/64, NTOK/16), b128, 0, stream>>>(mixed, sg2_w2, sg2_b2, u2, NTOK, D_, D_);
  swiglu_kernel<<<(NTOK*D_ + 255)/256, b256, 0, stream>>>(g2, u2, nullptr, wa, NTOK*D_);
  final_kernel<<<NTOK, w32, 0, stream>>>(x1, wa, eo, topw, xout, cons);
}
